// RawGlobalPointer_5884105196192
// MI455X (gfx1250) — matrix operand order repaired and checked by reading, not compiled, not run on hardware
//
#include <hip/hip_runtime.h>
#include <hip/hip_bf16.h>

// ---- problem constants (from reference) ----
#define BATCH   16
#define S_LEN   512
#define HDIM    768
#define NHEAD   9
#define DH      64
#define N1      1152            // NHEAD * DH * 2
#define NEGC    1000000000000.0f

typedef __attribute__((ext_vector_type(16))) __bf16 v16bf;
typedef __attribute__((ext_vector_type(8)))  float  v8f;
typedef __attribute__((ext_vector_type(4)))  int    v4i;

#define GLOBAL_AS __attribute__((address_space(1)))
#define LDS_AS    __attribute__((address_space(3)))

#if __has_builtin(__builtin_amdgcn_global_load_async_to_lds_b128)
#define HAVE_ASYNC 1
#else
#define HAVE_ASYNC 0
#endif

union Frag {
    v16bf          bf;
    unsigned short u[16];
    uint4          q[2];
};
static_assert(sizeof(Frag) == 32, "frag size");

// f32 -> bf16, round-half-up (cheap: one add + shift)
__device__ __forceinline__ unsigned short f2bf(float x) {
    return (unsigned short)((__float_as_uint(x) + 0x8000u) >> 16);
}

// two f32 -> packed bf16x2 in one dword: 2 adds + 1 v_perm_b32
__device__ __forceinline__ unsigned pack_bf16(float lo, float hi_) {
    unsigned a = __float_as_uint(hi_) + 0x8000u;   // -> high half
    unsigned b = __float_as_uint(lo)  + 0x8000u;   // -> low half
    // result = { a[31:16], b[31:16] }
    return __builtin_amdgcn_perm(a, b, 0x07060302u);
}

// =====================================================================
// Kernel 1: out = X @ W + b, split q/k, RoPE, pack bf16 into ws.
// X: (8192 x 768) f32, W: (768 x 1152) f32.  Block tile 64x128,
// double-buffered bf16 LDS tiles, global loads pipelined one k-tile
// ahead in registers.
// =====================================================================
struct StageRegs { float4 a[2]; float4 b[4]; };

__device__ __forceinline__ void load_tiles(
    const float* __restrict__ X, const float* __restrict__ W,
    int rowBase, int colBase, int k0, int tid, StageRegs& rg)
{
    // A: 64 rows x 32 k (float4 per thread x2)
#pragma unroll
    for (int i = 0; i < 2; ++i) {
        int idx = tid + i * 256;            // 0..511
        int r   = idx >> 3;                 // row 0..63
        int c4  = idx & 7;                  // float4 slot
        const float* gp = X + (size_t)(rowBase + r) * HDIM + k0 + c4 * 4;
        __builtin_prefetch(gp + 64, 0, 1);  // two k-tiles ahead
        rg.a[i] = *(const float4*)gp;
    }
    // B: 32(k) x 128(n) (float4 per thread x4)
#pragma unroll
    for (int i = 0; i < 4; ++i) {
        int idx = tid + i * 256;            // 0..1023
        int r   = idx >> 5;                 // k row 0..31
        int c4  = idx & 31;                 // col group
        const float* gp = W + (size_t)(k0 + r) * N1 + colBase + c4 * 4;
        __builtin_prefetch(gp + 64 * N1, 0, 1);
        rg.b[i] = *(const float4*)gp;
    }
}

__device__ __forceinline__ void store_tiles(
    const StageRegs& rg, int tid,
    unsigned short* ldsA, unsigned short* ldsB)
{
#pragma unroll
    for (int i = 0; i < 2; ++i) {
        int idx = tid + i * 256;
        int r   = idx >> 3;
        int c4  = idx & 7;
        uint2 st;
        st.x = pack_bf16(rg.a[i].x, rg.a[i].y);
        st.y = pack_bf16(rg.a[i].z, rg.a[i].w);
        *(uint2*)(&ldsA[r * 40 + c4 * 4]) = st;
    }
#pragma unroll
    for (int i = 0; i < 4; ++i) {
        int idx = tid + i * 256;
        int r   = idx >> 5;
        int c4  = idx & 31;
        int c   = c4 * 4;
        ldsB[(c + 0) * 40 + r] = f2bf(rg.b[i].x);
        ldsB[(c + 1) * 40 + r] = f2bf(rg.b[i].y);
        ldsB[(c + 2) * 40 + r] = f2bf(rg.b[i].z);
        ldsB[(c + 3) * 40 + r] = f2bf(rg.b[i].w);
    }
}

__global__ __launch_bounds__(256)
void proj_rope_kernel(const float* __restrict__ X,
                      const float* __restrict__ W,
                      const float* __restrict__ bias,
                      unsigned short* __restrict__ Qws,
                      unsigned short* __restrict__ Kws)
{
    const int tid  = threadIdx.x;
    const int lane = tid & 31;
    const int wave = tid >> 5;          // 0..7
    const int wm   = wave >> 2;         // 0..1  (32 rows each)
    const int wn   = wave & 3;          // 0..3  (32 cols each)
    const int lr   = lane & 15;
    const int hi   = lane >> 4;

    const int rowBase = blockIdx.x * 64;    // 0..8128
    const int colBase = blockIdx.y * 128;   // 0..1024

    __shared__ unsigned short ldsA[2][64 * 40];    // bf16, padded stride 40
    __shared__ unsigned short ldsB[2][128 * 40];   // bf16, [col][k]

    v8f acc[2][2];
    const v8f vzero = {0.f,0.f,0.f,0.f,0.f,0.f,0.f,0.f};
#pragma unroll
    for (int i = 0; i < 2; ++i)
#pragma unroll
        for (int j = 0; j < 2; ++j) acc[i][j] = vzero;

    // prologue: tile 0 into LDS, tile 1 into registers
    {
        StageRegs r0;
        load_tiles(X, W, rowBase, colBase, 0, tid, r0);
        store_tiles(r0, tid, ldsA[0], ldsB[0]);
    }
    StageRegs rg;
    load_tiles(X, W, rowBase, colBase, 32, tid, rg);
    __syncthreads();

#pragma unroll 2
    for (int kt = 0; kt < HDIM / 32; ++kt) {
        const int cur = kt & 1;
        // ---- build fragments: straight ds_load_b128 (already bf16)
        Frag afr[2], bfr[2];
#pragma unroll
        for (int tm = 0; tm < 2; ++tm) {
            int row = wm * 32 + tm * 16 + lr;
            const unsigned short* src = &ldsA[cur][row * 40];
            afr[tm].q[0] = *(const uint4*)(src + hi * 16);       // k = hi*16 ..
            afr[tm].q[1] = *(const uint4*)(src + hi * 16 + 8);   // k = hi*16+8 ..
        }
#pragma unroll
        for (int tn = 0; tn < 2; ++tn) {
            int col = wn * 32 + tn * 16 + lr;
            const unsigned short* src = &ldsB[cur][col * 40 + hi * 16];
            bfr[tn].q[0] = *(const uint4*)(src);
            bfr[tn].q[1] = *(const uint4*)(src + 8);
        }
#pragma unroll
        for (int tm = 0; tm < 2; ++tm)
#pragma unroll
            for (int tn = 0; tn < 2; ++tn)
                acc[tm][tn] = __builtin_amdgcn_wmma_f32_16x16x32_bf16(
                    false, afr[tm].bf, false, bfr[tn].bf,
                    (short)0, acc[tm][tn], false, false);

        // ---- drain the register tile into the other buffer, then
        //      issue global loads for the tile after next
        if (kt < HDIM / 32 - 1) {
            store_tiles(rg, tid, ldsA[cur ^ 1], ldsB[cur ^ 1]);
            if (kt < HDIM / 32 - 2)
                load_tiles(X, W, rowBase, colBase, 32 * (kt + 2), tid, rg);
        }
        __syncthreads();
    }

    // ---- epilogue: bias + RoPE + paired bf16 stores to workspace
#pragma unroll
    for (int tm = 0; tm < 2; ++tm) {
#pragma unroll
        for (int tn = 0; tn < 2; ++tn) {
            int col    = colBase + wn * 32 + tn * 16 + lr;   // 0..1151
            float bv   = bias[col];
            int head   = col >> 7;                           // / 128
            int within = col & 127;
            int d      = within & 63;
            bool isQ   = within < 64;
            int  fi    = d >> 1;
            float inv_freq = exp2f(-0.4152410118f * (float)fi); // 10000^(-fi/32)
            unsigned short* wsbase = isQ ? Qws : Kws;
#pragma unroll
            for (int v = 0; v < 8; ++v) {
                int m    = v + 8 * hi;
                int grow = rowBase + wm * 32 + tm * 16 + m;  // 0..8191
                int s    = grow & (S_LEN - 1);
                int bb   = grow >> 9;
                float x  = acc[tm][tn][v] + bv;
                // even/odd RoPE partner lives in lane^1 (col^1)
                float xp = __shfl_xor(x, 1, 32);
                float sn, cs;
                __sincosf((float)s * inv_freq, &sn, &cs);   // fast intrinsic
                float rot = (d & 1) ? (x * cs + xp * sn)
                                    : (x * cs - xp * sn);
                float rp  = __shfl_xor(rot, 1, 32);
                if ((lr & 1) == 0) {  // even lane stores the (d, d+1) pair
                    size_t dstRow = ((size_t)(bb * NHEAD + head) * S_LEN + s);
                    *(unsigned*)(wsbase + dstRow * DH + d) = pack_bf16(rot, rp);
                }
            }
        }
    }
}

// =====================================================================
// Kernel 2: logits[bt][m][n] = (q[m]·k[n]) masked, / 8.
// Per (b,t): 512x512x64 GEMM.  Block tile 64x128.
// grid = (8 m-tiles, 4 n-tiles, 144 bt)
// =====================================================================
__global__ __launch_bounds__(256)
void attn_logits_kernel(const unsigned short* __restrict__ Qws,
                        const unsigned short* __restrict__ Kws,
                        const int* __restrict__ amask,
                        float* __restrict__ out)
{
    const int tid  = threadIdx.x;
    const int lane = tid & 31;
    const int wave = tid >> 5;
    const int wm   = wave >> 2;
    const int wn   = wave & 3;
    const int lr   = lane & 15;
    const int hi   = lane >> 4;

    const int bt = blockIdx.z;          // b*NHEAD + t
    const int bb = bt / NHEAD;
    const int mBase = blockIdx.x * 64;
    const int nBase = blockIdx.y * 128;

    __shared__ unsigned short ldsQ[64  * 72];   // [row][d], stride 72
    __shared__ unsigned short ldsK[128 * 72];

    const unsigned short* Qbase = Qws + (size_t)bt * S_LEN * DH + (size_t)mBase * DH;
    const unsigned short* Kbase = Kws + (size_t)bt * S_LEN * DH + (size_t)nBase * DH;

#if HAVE_ASYNC
    // CDNA5 async copy: global -> LDS without touching VGPR data
#pragma unroll
    for (int i = 0; i < 2; ++i) {
        int idx = tid + i * 256;
        int r = idx >> 3, c = idx & 7;
        __builtin_amdgcn_global_load_async_to_lds_b128(
            (GLOBAL_AS v4i*)(Qbase + (size_t)r * DH + c * 8),
            (LDS_AS v4i*)(&ldsQ[r * 72 + c * 8]), 0, 0);
    }
#pragma unroll
    for (int i = 0; i < 4; ++i) {
        int idx = tid + i * 256;
        int r = idx >> 3, c = idx & 7;
        __builtin_amdgcn_global_load_async_to_lds_b128(
            (GLOBAL_AS v4i*)(Kbase + (size_t)r * DH + c * 8),
            (LDS_AS v4i*)(&ldsK[r * 72 + c * 8]), 0, 0);
    }
    asm volatile("s_wait_asynccnt 0" ::: "memory");
#else
#pragma unroll
    for (int i = 0; i < 2; ++i) {
        int idx = tid + i * 256;
        int r = idx >> 3, c = idx & 7;
        uint4 v = *(const uint4*)(Qbase + (size_t)r * DH + c * 8);
        *(uint4*)(&ldsQ[r * 72 + c * 8]) = v;
    }
#pragma unroll
    for (int i = 0; i < 4; ++i) {
        int idx = tid + i * 256;
        int r = idx >> 3, c = idx & 7;
        uint4 v = *(const uint4*)(Kbase + (size_t)r * DH + c * 8);
        *(uint4*)(&ldsK[r * 72 + c * 8]) = v;
    }
#endif
    __syncthreads();

    v8f acc[2][2];
    const v8f vzero = {0.f,0.f,0.f,0.f,0.f,0.f,0.f,0.f};
#pragma unroll
    for (int i = 0; i < 2; ++i)
#pragma unroll
        for (int j = 0; j < 2; ++j) acc[i][j] = vzero;

#pragma unroll
    for (int k0 = 0; k0 < DH; k0 += 32) {
        Frag afr[2], bfr[2];
#pragma unroll
        for (int tm = 0; tm < 2; ++tm) {
            int row = wm * 32 + tm * 16 + lr;
            const unsigned short* src = &ldsQ[row * 72 + k0];
            afr[tm].q[0] = *(const uint4*)(src + hi * 16);       // k = hi*16 .. +7
            afr[tm].q[1] = *(const uint4*)(src + hi * 16 + 8);   // k = hi*16+8 .. +15
        }
#pragma unroll
        for (int tn = 0; tn < 2; ++tn) {
            int col = wn * 32 + tn * 16 + lr;                    // key row in K
            const unsigned short* src = &ldsK[col * 72 + k0 + hi * 16];
            bfr[tn].q[0] = *(const uint4*)(src);                 // k = hi*16 .. +7
            bfr[tn].q[1] = *(const uint4*)(src + 8);             // k = hi*16+8 .. +15
        }
#pragma unroll
        for (int tm = 0; tm < 2; ++tm)
#pragma unroll
            for (int tn = 0; tn < 2; ++tn)
                acc[tm][tn] = __builtin_amdgcn_wmma_f32_16x16x32_bf16(
                    false, afr[tm].bf, false, bfr[tn].bf,
                    (short)0, acc[tm][tn], false, false);
    }

    // ---- epilogue: padding mask, strict-causal mask, scale 1/8
#pragma unroll
    for (int tm = 0; tm < 2; ++tm) {
#pragma unroll
        for (int tn = 0; tn < 2; ++tn) {
            int n = nBase + wn * 32 + tn * 16 + lr;
            float pad = (float)amask[bb * S_LEN + n];
#pragma unroll
            for (int v = 0; v < 8; ++v) {
                int m = mBase + wm * 32 + tm * 16 + v + 8 * hi;
                float l = acc[tm][tn][v];
                l = l * pad - (1.0f - pad) * NEGC;
                if (m > n) l -= NEGC;
                out[((size_t)(bt * S_LEN + m)) * S_LEN + n] = l * 0.125f;
            }
        }
    }
}

// =====================================================================
extern "C" void kernel_launch(void* const* d_in, const int* in_sizes, int n_in,
                              void* d_out, int out_size, void* d_ws, size_t ws_size,
                              hipStream_t stream)
{
    const float* X     = (const float*)d_in[0];   // (16,512,768) f32
    const int*   amask = (const int*)  d_in[1];   // (16,512) i32
    const float* W     = (const float*)d_in[2];   // (768,1152) f32
    const float* bias  = (const float*)d_in[3];   // (1152,) f32
    float*       out   = (float*)d_out;           // (16,9,512,512) f32

    unsigned short* Qws = (unsigned short*)d_ws;
    unsigned short* Kws = Qws + (size_t)BATCH * NHEAD * S_LEN * DH;   // +4.7M elems

    dim3 g1(128, 9, 1);          // (8192/64) x (1152/128)
    proj_rope_kernel<<<g1, 256, 0, stream>>>(X, W, bias, Qws, Kws);

    dim3 g2(8, 4, BATCH * NHEAD);  // (512/64) x (512/128) x 144
    attn_logits_kernel<<<g2, 256, 0, stream>>>(Qws, Kws, amask, out);
}
